// DCFNet_69647189671988
// MI455X (gfx1250) — matrix operand order repaired and checked by reading, not compiled, not run on hardware
//
#include <hip/hip_runtime.h>

typedef __attribute__((ext_vector_type(16))) _Float16 v16h;
typedef __attribute__((ext_vector_type(8)))  _Float16 v8h;
typedef __attribute__((ext_vector_type(8)))  float    v8f;

#define NS   16
#define CIN  3
#define CF   32
#define CROP 256
#define KXH  129   // CROP/2 + 1

struct cpx { float x, y; };
__device__ inline cpx cadd(cpx a, cpx b){ return {a.x+b.x, a.y+b.y}; }
__device__ inline cpx csub(cpx a, cpx b){ return {a.x-b.x, a.y-b.y}; }
__device__ inline cpx cmul(cpx a, cpx b){ return {a.x*b.x - a.y*b.y, a.x*b.y + a.y*b.x}; }

// Stockham radix-2 FFT, N=256, 128 threads, ping-pong LDS buffers.
// sgn = -1 forward, +1 inverse (unnormalized). Result ends in A.
__device__ inline void fft256(cpx* A, cpx* B, int t, float sgn) {
  cpx* cur = A; cpx* nxt = B;
  for (int p = 1; p < 256; p <<= 1) {
    __syncthreads();
    int k = t & (p - 1);
    cpx u = cur[t];
    cpx v = cur[t + 128];
    float ang = sgn * 3.14159265358979f * (float)k / (float)p;
    cpx w = { __cosf(ang), __sinf(ang) };
    cpx vw = cmul(v, w);
    nxt[2*t - k]     = cadd(u, vw);
    nxt[2*t - k + p] = csub(u, vw);
    cpx* tmp = cur; cur = nxt; nxt = tmp;
  }
  __syncthreads();
}

// ---------------- weight prep: w2 (OIHW f32) -> w2t[tap][cout][cin] f16 ----
__global__ __launch_bounds__(256) void prep_w2t_kernel(const float* __restrict__ w2,
                                                       _Float16* __restrict__ w2t) {
  for (int idx = threadIdx.x; idx < 9 * CF * CF; idx += 256) {
    int tap = idx / (CF * CF);
    int r   = idx % (CF * CF);
    int co  = r / CF;
    int ci  = r % CF;
    int ky = tap / 3, kx = tap % 3;
    w2t[idx] = (_Float16)w2[((co * CF + ci) * 3 + ky) * 3 + kx];
  }
}

// ---------------- conv1 + ReLU -> f1 in NHWC f16 ---------------------------
__global__ __launch_bounds__(256) void conv1_kernel(const float* __restrict__ z,
                                                    const float* __restrict__ w1,
                                                    const float* __restrict__ b1,
                                                    _Float16* __restrict__ f1t) {
  __shared__ float w1s[CF * CIN * 9];
  __shared__ float b1s[CF];
  int t = threadIdx.x;
  for (int i = t; i < CF * CIN * 9; i += 256) w1s[i] = w1[i];
  if (t < CF) b1s[t] = b1[t];
  __syncthreads();

  int bb = blockIdx.x;
  int n = bb >> 8, y = bb & 255, x = t;

  float in[CIN][3][3];
  #pragma unroll
  for (int ci = 0; ci < CIN; ++ci)
    #pragma unroll
    for (int ky = 0; ky < 3; ++ky)
      #pragma unroll
      for (int kx = 0; kx < 3; ++kx) {
        int iy = y + ky - 1, ix = x + kx - 1;
        bool ok = (iy >= 0) & (iy < CROP) & (ix >= 0) & (ix < CROP);
        in[ci][ky][kx] = ok ? z[((size_t)(n * CIN + ci) * CROP + iy) * CROP + ix] : 0.0f;
      }

  _Float16 h[CF];
  #pragma unroll
  for (int c = 0; c < CF; ++c) {
    float acc = b1s[c];
    #pragma unroll
    for (int ci = 0; ci < CIN; ++ci)
      #pragma unroll
      for (int ky = 0; ky < 3; ++ky)
        #pragma unroll
        for (int kx = 0; kx < 3; ++kx)
          acc = fmaf(w1s[((c * CIN + ci) * 3 + ky) * 3 + kx], in[ci][ky][kx], acc);
    h[c] = (_Float16)fmaxf(acc, 0.0f);
  }

  size_t base = ((size_t)(n * CROP + y) * CROP + x) * CF;
  v8h* dst = (v8h*)(f1t + base);
  #pragma unroll
  for (int i = 0; i < 4; ++i) {
    v8h hv;
    #pragma unroll
    for (int j = 0; j < 8; ++j) hv[j] = h[8 * i + j];
    dst[i] = hv;
  }
}

// ---------------- conv2 (WMMA) + bias + LRN + cos_window + channel-sum -----
__global__ __launch_bounds__(256) void conv2_kernel(const _Float16* __restrict__ f1t,
                                                    const _Float16* __restrict__ w2t,
                                                    const float* __restrict__ b2,
                                                    const float* __restrict__ cosw,
                                                    float* __restrict__ sbuf) {
  __shared__ float tileD[8][16][CF];   // 16 KB
  int t    = threadIdx.x;
  int wave = t >> 5;
  int lane = t & 31;
  int bb   = blockIdx.x;
  int n    = bb >> 9;
  int r9   = bb & 511;
  int y    = r9 >> 1;
  int half = r9 & 1;
  int x0   = half * 128 + wave * 16;   // 16-pixel strip base for this wave
  int M    = lane & 15;
  int g    = lane >> 4;

  union U { v16h v; v8h h[2]; };
  v8h zz = {};
  v8f acc0 = {}; v8f acc1 = {};

  for (int dy = 0; dy < 3; ++dy) {
    int iy = y + dy - 1;
    if (iy < 0 || iy >= CROP) continue;           // wave-uniform branch
    #pragma unroll
    for (int dx = 0; dx < 3; ++dx) {
      int ix = x0 + M + dx - 1;
      bool inb = (ix >= 0) & (ix < CROP);
      // A operand: 16 pixels x 32 in-channels (f16, NHWC => two b128 per lane)
      U a;
      if (inb) {
        const _Float16* p = f1t + ((size_t)(n * CROP + iy) * CROP + ix) * CF;
        a.h[0] = *(const v8h*)(p + 16 * g);
        a.h[1] = *(const v8h*)(p + 16 * g + 8);
      } else {
        a.h[0] = zz; a.h[1] = zz;
      }
      int tap = dy * 3 + dx;
      const _Float16* wp = w2t + (size_t)tap * (CF * CF);
      {
        U b;
        const _Float16* q = wp + (size_t)(M) * CF + 16 * g;        // cout 0..15
        b.h[0] = *(const v8h*)q;
        b.h[1] = *(const v8h*)(q + 8);
        acc0 = __builtin_amdgcn_wmma_f32_16x16x32_f16(false, a.v, false, b.v,
                                                      (short)0, acc0, false, false);
      }
      {
        U b;
        const _Float16* q = wp + (size_t)(16 + M) * CF + 16 * g;   // cout 16..31
        b.h[0] = *(const v8h*)q;
        b.h[1] = *(const v8h*)(q + 8);
        acc1 = __builtin_amdgcn_wmma_f32_16x16x32_f16(false, a.v, false, b.v,
                                                      (short)0, acc1, false, false);
      }
    }
  }

  // D layout: lane holds column N=lane%16, rows M = r + 8*(lane/16)
  int mrow = (lane >> 4) * 8;
  int nn   = lane & 15;
  #pragma unroll
  for (int r = 0; r < 8; ++r) {
    tileD[wave][mrow + r][nn]      = acc0[r];
    tileD[wave][mrow + r][nn + 16] = acc1[r];
  }
  __syncthreads();

  // bias + LRN(5, alpha=1e-4, beta=0.75, k=1) + cos window + channel sum
  if (t < 128) {
    int strip = t >> 4, m = t & 15;
    int gx = half * 128 + strip * 16 + m;
    float vals[CF], sq[CF];
    #pragma unroll
    for (int c = 0; c < CF; ++c) {
      vals[c] = tileD[strip][m][c] + b2[c];
      sq[c]   = vals[c] * vals[c];
    }
    float ssum = 0.0f;
    #pragma unroll
    for (int c = 0; c < CF; ++c) {
      int lo = c - 2 < 0 ? 0 : c - 2;
      int hi = c + 2 > CF - 1 ? CF - 1 : c + 2;
      float w = 0.0f;
      for (int j = lo; j <= hi; ++j) w += sq[j];
      ssum += vals[c] * __powf(1.0f + 2e-5f * w, -0.75f);
    }
    float cw = cosw[y * CROP + gx];
    sbuf[((size_t)n * CROP + y) * CROP + gx] = ssum * cw;
  }
}

// ---------------- forward row rFFT: s[n][y][:] -> Sx[n][kx][y] -------------
__global__ __launch_bounds__(128) void fftx_kernel(const float* __restrict__ sbuf,
                                                   cpx* __restrict__ Sx) {
  __shared__ cpx A[256], B[256];
  int t = threadIdx.x, bb = blockIdx.x;
  int n = bb >> 8, y = bb & 255;
  const float* row = sbuf + ((size_t)n * CROP + y) * CROP;
  A[t]       = { row[t],       0.0f };
  A[t + 128] = { row[t + 128], 0.0f };
  fft256(A, B, t, -1.0f);
  Sx[((size_t)n * KXH + t) * CROP + y] = A[t];
  if (t == 0) Sx[((size_t)n * KXH + 128) * CROP + y] = A[128];
}

// ---------------- column FFT, multiply by wc=conj-style, inverse column ----
__global__ __launch_bounds__(128) void ffty_kernel(cpx* __restrict__ Sx,
                                                   const float* __restrict__ wf) {
  __shared__ cpx A[256], B[256];
  int t = threadIdx.x, bb = blockIdx.x;
  int n = bb / KXH, kx = bb % KXH;
  cpx* col = Sx + ((size_t)n * KXH + kx) * CROP;
  A[t]       = col[t];
  A[t + 128] = col[t + 128];
  fft256(A, B, t, -1.0f);
  // wc = wf[0,1,ky,kx,0] - i*wf[0,1,ky,kx,1]
  #pragma unroll
  for (int i = 0; i < 2; ++i) {
    int ky = t + i * 128;
    size_t wi = ((size_t)(CROP + ky) * KXH + kx) * 2;   // channel 1 of 32
    float wr = wf[wi], wim = wf[wi + 1];
    cpx X = A[ky];
    A[ky] = { X.x * wr + X.y * wim, X.y * wr - X.x * wim };
  }
  fft256(A, B, t, +1.0f);   // has leading __syncthreads per stage
  col[t]       = A[t];
  col[t + 128] = A[t + 128];
}

// ---------------- inverse row FFT with hermitian reconstruction -> out -----
__global__ __launch_bounds__(128) void ifftx_kernel(const cpx* __restrict__ Sx,
                                                    float* __restrict__ out) {
  __shared__ cpx A[256], B[256];
  int t = threadIdx.x, bb = blockIdx.x;
  int n = bb >> 8, y = bb & 255;
  const cpx* base = Sx + (size_t)n * KXH * CROP;
  A[t] = base[(size_t)t * CROP + y];
  cpx v;
  if (t == 0) {
    v = base[(size_t)128 * CROP + y];
  } else {
    v = base[(size_t)(128 - t) * CROP + y];
    v.y = -v.y;                                   // X[k] = conj(X[256-k])
  }
  A[t + 128] = v;
  fft256(A, B, t, +1.0f);
  const float scale = 1.0f / 65536.0f;            // irfft2 normalization
  float* orow = out + ((size_t)n * CROP + y) * CROP;
  orow[t]       = A[t].x * scale;
  orow[t + 128] = A[t + 128].x * scale;
}

extern "C" void kernel_launch(void* const* d_in, const int* in_sizes, int n_in,
                              void* d_out, int out_size, void* d_ws, size_t ws_size,
                              hipStream_t stream) {
  const float* z  = (const float*)d_in[0];
  const float* w1 = (const float*)d_in[1];
  const float* b1 = (const float*)d_in[2];
  const float* w2 = (const float*)d_in[3];
  const float* b2 = (const float*)d_in[4];
  const float* cw = (const float*)d_in[5];
  const float* wf = (const float*)d_in[6];
  float* out = (float*)d_out;

  char* ws = (char*)d_ws;
  _Float16* f1t = (_Float16*)ws;                                  // 64 MB
  size_t off = (size_t)NS * CROP * CROP * CF * sizeof(_Float16);
  float* sbuf = (float*)(ws + off);                               // 4 MB
  off += (size_t)NS * CROP * CROP * sizeof(float);
  cpx* Sx = (cpx*)(ws + off);                                     // ~4 MB
  off += (size_t)NS * KXH * CROP * sizeof(cpx);
  _Float16* w2t = (_Float16*)(ws + off);                          // 18 KB

  prep_w2t_kernel<<<1, 256, 0, stream>>>(w2, w2t);
  conv1_kernel<<<NS * CROP, 256, 0, stream>>>(z, w1, b1, f1t);
  conv2_kernel<<<NS * 512, 256, 0, stream>>>(f1t, w2t, b2, cw, sbuf);
  fftx_kernel<<<NS * CROP, 128, 0, stream>>>(sbuf, Sx);
  ffty_kernel<<<NS * KXH, 128, 0, stream>>>(Sx, wf);
  ifftx_kernel<<<NS * CROP, 128, 0, stream>>>(Sx, out);
}
